// facenet_12790412608056
// MI455X (gfx1250) — matrix operand order repaired and checked by reading, not compiled, not run on hardware
//
#include <hip/hip_runtime.h>
#include <hip/hip_bf16.h>

typedef __attribute__((ext_vector_type(16))) _Float16 v16h;
typedef __attribute__((ext_vector_type(8)))  float    v8f;

#define NNODES 100000
#define NEDGES 1600000
#define HDIM   64
#define BN_EPS 1e-5f
#define SLOPE  0.01f

// ---------------------------------------------------------------------------
// Edge scatter: msg[dst] += h[src] (64ch), deg[dst] += 1.
// 16 threads per edge, float4 gather (L2-resident), 4 f32 atomics per thread.
// ---------------------------------------------------------------------------
__global__ __launch_bounds__(256) void scatter_kernel(
    const float* __restrict__ h, const int* __restrict__ src,
    const int* __restrict__ dst, float* __restrict__ msg,
    float* __restrict__ deg)
{
    long long t = (long long)blockIdx.x * blockDim.x + threadIdx.x;
    int e = (int)(t >> 4);
    if (e >= NEDGES) return;
    int part = (int)(t & 15);
    int s = src[e], d = dst[e];
    const float4 v = *(const float4*)(h + (long long)s * HDIM + part * 4);
    float* mp = msg + (long long)d * HDIM + part * 4;
    atomicAdd(mp + 0, v.x);
    atomicAdd(mp + 1, v.y);
    atomicAdd(mp + 2, v.z);
    atomicAdd(mp + 3, v.w);
    if (part == 0) atomicAdd(deg + d, 1.0f);
}

// ---------------------------------------------------------------------------
// SAGE linear:  out = h @ Ws^T + (msg/max(deg,1)) @ Wn^T + b
// One block = 16 output rows; 4 waves, wave w owns N-tile [16w,16w+16).
// K=64 per operand -> 2x v_wmma_f32_16x16x32_f16 per operand, f32 accum.
// ---------------------------------------------------------------------------
__global__ __launch_bounds__(128) void sage_gemm_kernel(
    const float* __restrict__ h,    const float* __restrict__ msg,
    const float* __restrict__ deg,  const float* __restrict__ Wself,
    const float* __restrict__ Wneigh, const float* __restrict__ bias,
    float* __restrict__ out)
{
    __shared__ _Float16 sW[2][HDIM][HDIM + 2];  // [self|neigh][outCh][inCh], padded
    __shared__ float    sA[2][16][HDIM + 4];    // [self|neigh][row][ch], padded

    const int tid  = threadIdx.x;   // 0..127
    const int wave = tid >> 5;      // 0..3
    const int lane = tid & 31;
    const int l16  = lane & 15;
    const int half = lane >> 4;     // 0: lanes 0-15, 1: lanes 16-31
    const int row0 = blockIdx.x * 16;

    // Stage weights as f16 (16 KB) -- L2-resident, tiny.
    for (int i = tid; i < HDIM * HDIM; i += 128) {
        int r = i >> 6, c = i & 63;
        sW[0][r][c] = (_Float16)Wself[i];
        sW[1][r][c] = (_Float16)Wneigh[i];
    }
    // Stage 16 rows of self features and mean-aggregated neighbor features.
    for (int i = tid; i < 16 * HDIM; i += 128) {
        int r = i >> 6, c = i & 63;
        int gr = row0 + r;
        float hv = 0.f, nv = 0.f;
        if (gr < NNODES) {
            hv = h[(long long)gr * HDIM + c];
            nv = msg[(long long)gr * HDIM + c] / fmaxf(deg[gr], 1.0f);
        }
        sA[0][r][c] = hv;
        sA[1][r][c] = nv;
    }
    __syncthreads();

    const int nb = wave * 16;       // N-tile base
    v8f acc = {};

    #pragma unroll
    for (int srcm = 0; srcm < 2; ++srcm) {
        #pragma unroll
        for (int kb = 0; kb < HDIM; kb += 32) {
            // A fragment (16x32 f16): lane half selects K-subgroups per ISA layout.
            v16h a;
            const int ka = kb + half * 8;
            #pragma unroll
            for (int j = 0; j < 8; ++j) {
                a[j]     = (_Float16)sA[srcm][l16][ka + j];
                a[8 + j] = (_Float16)sA[srcm][l16][ka + 16 + j];
            }
            // B fragment (32x16 f16): B[k][n] = W[n][k]; col = nb+l16.
            v16h b;
            const int kbb = kb + half * 8;
            #pragma unroll
            for (int j = 0; j < 8; ++j) {
                b[j]     = sW[srcm][nb + l16][kbb + j];
                b[8 + j] = sW[srcm][nb + l16][kbb + 16 + j];
            }

            acc = __builtin_amdgcn_wmma_f32_16x16x32_f16(
                false, a, false, b, (short)0, acc, false, false);
        }
    }

    // D layout: VGPR r -> row r + 8*half, col = nb + l16.
    const int col = nb + l16;
    const float bv = bias[col];
    #pragma unroll
    for (int r = 0; r < 8; ++r) {
        int gr = row0 + r + half * 8;
        if (gr < NNODES)
            out[(long long)gr * HDIM + col] = acc[r] + bv;
    }
}

// ---------------------------------------------------------------------------
// Per-channel sum / sumsq over all N rows (for BatchNorm).
// stats[0..63] = sum, stats[64..127] = sumsq.
// ---------------------------------------------------------------------------
__global__ __launch_bounds__(256) void stats_kernel(
    const float* __restrict__ pre, float* __restrict__ stats)
{
    __shared__ float ssum[HDIM];
    __shared__ float ssq[HDIM];
    const int tid = threadIdx.x;
    if (tid < HDIM) { ssum[tid] = 0.f; ssq[tid] = 0.f; }
    __syncthreads();

    const int c = tid & 63;   // stride is a multiple of 64 -> channel invariant
    float s = 0.f, q = 0.f;
    const long long total  = (long long)NNODES * HDIM;
    const long long stride = (long long)gridDim.x * blockDim.x;
    for (long long i = (long long)blockIdx.x * blockDim.x + tid; i < total; i += stride) {
        float v = pre[i];
        s += v;
        q += v * v;
    }
    atomicAdd(&ssum[c], s);
    atomicAdd(&ssq[c], q);
    __syncthreads();
    if (tid < HDIM) {
        atomicAdd(&stats[tid], ssum[tid]);
        atomicAdd(&stats[HDIM + tid], ssq[tid]);
    }
}

// stats[128..191] = mu, stats[192..255] = inv_std
__global__ __launch_bounds__(64) void bn_finalize_kernel(float* __restrict__ stats)
{
    int c = threadIdx.x;
    float mu  = stats[c] / (float)NNODES;
    float var = stats[HDIM + c] / (float)NNODES - mu * mu;
    stats[128 + c] = mu;
    stats[192 + c] = rsqrtf(var + BN_EPS);
}

// out = leaky_relu((pre - mu) * inv_std * gamma + beta)
__global__ __launch_bounds__(256) void bn_apply_kernel(
    const float* __restrict__ pre, const float* __restrict__ stats,
    const float* __restrict__ gamma, const float* __restrict__ beta,
    float* __restrict__ out)
{
    long long i = (long long)blockIdx.x * blockDim.x + threadIdx.x;
    if (i >= (long long)NNODES * HDIM) return;
    int c = (int)(i & 63);
    float v = (pre[i] - stats[128 + c]) * stats[192 + c] * gamma[c] + beta[c];
    out[i] = (v > 0.f) ? v : SLOPE * v;
}

// ---------------------------------------------------------------------------
// Host-side orchestration
// ---------------------------------------------------------------------------
extern "C" void kernel_launch(void* const* d_in, const int* in_sizes, int n_in,
                              void* d_out, int out_size, void* d_ws, size_t ws_size,
                              hipStream_t stream)
{
    const float* x        = (const float*)d_in[0];
    const int*   eidx     = (const int*)d_in[1];
    const float* Wself1   = (const float*)d_in[2];
    const float* Wneigh1  = (const float*)d_in[3];
    const float* b1       = (const float*)d_in[4];
    const float* gamma1   = (const float*)d_in[5];
    const float* beta1    = (const float*)d_in[6];
    const float* Wself2   = (const float*)d_in[7];
    const float* Wneigh2  = (const float*)d_in[8];
    const float* b2       = (const float*)d_in[9];
    const float* gamma2   = (const float*)d_in[10];
    const float* beta2    = (const float*)d_in[11];
    float* outp = (float*)d_out;

    const int* src = eidx;           // edge_index[0]
    const int* dst = eidx + NEDGES;  // edge_index[1]

    // Workspace layout (bytes): [msg | deg | stats | h]
    const size_t MSG_BYTES   = (size_t)NNODES * HDIM * sizeof(float); // 25.6 MB
    const size_t DEG_BYTES   = (size_t)NNODES * sizeof(float);        // 0.4 MB
    const size_t STATS_BYTES = 256 * sizeof(float);
    char* ws = (char*)d_ws;
    float* msg   = (float*)(ws);
    float* deg   = (float*)(ws + MSG_BYTES);
    float* stats = (float*)(ws + MSG_BYTES + DEG_BYTES);
    float* hbuf  = (float*)(ws + MSG_BYTES + DEG_BYTES + STATS_BYTES);
    const size_t ZERO_BYTES = MSG_BYTES + DEG_BYTES + STATS_BYTES;

    const int scatterBlocks = (int)(((long long)NEDGES * 16 + 255) / 256);
    const int gemmBlocks    = (NNODES + 15) / 16;   // 6250
    const int applyBlocks   = (int)(((long long)NNODES * HDIM + 255) / 256);

    // ----- Layer 1 -----
    hipMemsetAsync(msg, 0, ZERO_BYTES, stream);
    scatter_kernel<<<scatterBlocks, 256, 0, stream>>>(x, src, dst, msg, deg);
    sage_gemm_kernel<<<gemmBlocks, 128, 0, stream>>>(x, msg, deg, Wself1, Wneigh1, b1, hbuf);
    stats_kernel<<<512, 256, 0, stream>>>(hbuf, stats);
    bn_finalize_kernel<<<1, 64, 0, stream>>>(stats);
    bn_apply_kernel<<<applyBlocks, 256, 0, stream>>>(hbuf, stats, gamma1, beta1, hbuf);

    // ----- Layer 2 -----
    hipMemsetAsync(msg, 0, ZERO_BYTES, stream);
    scatter_kernel<<<scatterBlocks, 256, 0, stream>>>(hbuf, src, dst, msg, deg);
    // In-place pre-BN write over hbuf is safe: each block stages its own 16
    // rows to LDS (all reads) before writing those same rows.
    sage_gemm_kernel<<<gemmBlocks, 128, 0, stream>>>(hbuf, msg, deg, Wself2, Wneigh2, b2, hbuf);
    stats_kernel<<<512, 256, 0, stream>>>(hbuf, stats);
    bn_finalize_kernel<<<1, 64, 0, stream>>>(stats);
    bn_apply_kernel<<<applyBlocks, 256, 0, stream>>>(hbuf, stats, gamma2, beta2, outp);
}
